// MSDeformAttn_32504312496244
// MI455X (gfx1250) — matrix operand order repaired and checked by reading, not compiled, not run on hardware
//
#include <hip/hip_runtime.h>
#include <hip/hip_bf16.h>

typedef _Float16 v8h  __attribute__((ext_vector_type(8)));
typedef _Float16 v16h __attribute__((ext_vector_type(16)));
typedef float    v8f  __attribute__((ext_vector_type(8)));

#define NB   2
#define LQ   4096
#define CDIM 256
#define MH   8
#define LL   4
#define PP   4
#define DH   32
#define LIN  74880   // 16*64*64 + 8*32*32 + 4*16*16 + 2*8*8

// ---------------------------------------------------------------------------
// fp32 -> f16 elementwise convert (grid-stride)
// ---------------------------------------------------------------------------
__global__ void f32_to_f16_kernel(const float* __restrict__ in,
                                  _Float16* __restrict__ out, int n) {
  int i = blockIdx.x * blockDim.x + threadIdx.x;
  int stride = gridDim.x * blockDim.x;
  for (; i < n; i += stride) out[i] = (_Float16)in[i];
}

// ---------------------------------------------------------------------------
// Weight (K, Ncols) fp32 row-major  ->  (Ncols, K) f16 row-major (K contiguous)
// so each lane's B fragment is a contiguous 32B load.
// ---------------------------------------------------------------------------
__global__ void wt_transpose_kernel(const float* __restrict__ w,
                                    _Float16* __restrict__ wt,
                                    int K, int Ncols) {
  int i = blockIdx.x * blockDim.x + threadIdx.x;
  int total = K * Ncols;
  if (i >= total) return;
  int n = i / K, k = i % K;
  wt[i] = (_Float16)w[(size_t)k * Ncols + n];
}

// ---------------------------------------------------------------------------
// WMMA GEMM, 2x2 register-blocked: each wave computes a 32x32 output block
// (4 x v_wmma_f32_16x16x32_f16 per K-step from 4 A + 4 B b128 loads).
// C(M,N) f32 = A(M,K) f16 row-major  x  Bt(N,K) f16 (pre-transposed) + bias(N)
// Fragment layouts (C/D as in CDNA5 ISA 7.12.2; A/B k order is this code's own, same for both):
//   A 16-bit 16x32: lane m=lane&15; 8 VGPRs hold K = kbase + (lane>>4)*16 .. +15
//   B 16-bit 32x16: lane n=lane&15; 8 VGPRs hold K = kbase + (lane>>4)*16 .. +15
//   C/D f32 16x16 : VGPR r -> M = r + (lane>>4)*8, N = lane&15
// ---------------------------------------------------------------------------
#define WMMA_F16(A_, B_, C_) \
  __builtin_amdgcn_wmma_f32_16x16x32_f16(false, (A_), false, (B_), (short)0, (C_), false, false)

__global__ __launch_bounds__(128)
void gemm_f16_wmma_kernel(const _Float16* __restrict__ A,
                          const _Float16* __restrict__ Bt,
                          const float* __restrict__ bias,
                          float* __restrict__ Cout,
                          int Mrows, int Ncols, int K) {
  const int wave = threadIdx.x >> 5;
  const int lane = threadIdx.x & 31;
  const int tM   = blockIdx.x * 4 + wave;   // 32-row block index
  const int tN   = blockIdx.y;              // 32-col block index
  if (tM * 32 >= Mrows) return;             // wave-uniform: EXEC stays all-1s

  const int hi  = lane >> 4;
  const int l15 = lane & 15;

  const _Float16* aptr0 = A  + (size_t)(tM * 32 + l15) * K + hi * 16;
  const _Float16* aptr1 = aptr0 + (size_t)16 * K;
  const _Float16* bptr0 = Bt + (size_t)(tN * 32 + l15) * K + hi * 16;
  const _Float16* bptr1 = bptr0 + (size_t)16 * K;

  v8f c00 = {}, c01 = {}, c10 = {}, c11 = {};
  for (int k0 = 0; k0 < K; k0 += 32) {
    v16h a0 = *(const v16h*)(aptr0 + k0);
    v16h a1 = *(const v16h*)(aptr1 + k0);
    v16h b0 = *(const v16h*)(bptr0 + k0);   // 32B contiguous
    v16h b1 = *(const v16h*)(bptr1 + k0);   // 32B contiguous
    c00 = WMMA_F16(a0, b0, c00);
    c01 = WMMA_F16(a0, b1, c01);
    c10 = WMMA_F16(a1, b0, c10);
    c11 = WMMA_F16(a1, b1, c11);
  }

  const float bv0 = bias[tN * 32 + l15];
  const float bv1 = bias[tN * 32 + 16 + l15];
  const int   row0 = tM * 32 + hi * 8;      // C/D: VGPR r -> M = row0 + r
  float* o0 = Cout + (size_t)row0 * Ncols + tN * 32 + l15;
  float* o1 = o0 + (size_t)16 * Ncols;
#pragma unroll
  for (int r = 0; r < 8; ++r) {
    o0[(size_t)r * Ncols]      = c00[r] + bv0;
    o0[(size_t)r * Ncols + 16] = c01[r] + bv1;
    o1[(size_t)r * Ncols]      = c10[r] + bv0;
    o1[(size_t)r * Ncols + 16] = c11[r] + bv1;
  }
}

// ---------------------------------------------------------------------------
// In-place softmax over rows of 32 (lane = element; exactly one wave32 / row)
// ---------------------------------------------------------------------------
__global__ void softmax32_kernel(float* __restrict__ x, int nrows) {
  int row  = blockIdx.x * (blockDim.x >> 5) + (threadIdx.x >> 5);
  int lane = threadIdx.x & 31;
  if (row >= nrows) return;
  float v = x[(size_t)row * 32 + lane];
  float m = v;
  for (int o = 16; o > 0; o >>= 1) m = fmaxf(m, __shfl_xor(m, o, 32));
  float e = __expf(v - m);
  float s = e;
  for (int o = 16; o > 0; o >>= 1) s += __shfl_xor(s, o, 32);
  x[(size_t)row * 32 + lane] = e / s;
}

// ---------------------------------------------------------------------------
// Trilinear sampling + attention-weighted sum.
// One wave per (n,q,m); lane = channel d in [0,32). value layout (N,Lin,M,Dh)
// so each corner read is 32 contiguous floats (one 128B line per wave).
// Writes f16 directly (consumed by the final WMMA GEMM).
// ---------------------------------------------------------------------------
__global__ __launch_bounds__(128)
void msdeform_sample_kernel(const float* __restrict__ refp,   // (N,Lq,L,3)
                            const float* __restrict__ off,    // (N,Lq,M,L,P,3)
                            const float* __restrict__ attn,   // (N,Lq,M,L*P) post-softmax
                            const float* __restrict__ value,  // (N,Lin,M,Dh)
                            _Float16* __restrict__ t2f16) {   // (N,Lq,M,Dh)
  const int gid  = blockIdx.x * 4 + (threadIdx.x >> 5);
  const int lane = threadIdx.x & 31;
  if (gid >= NB * LQ * MH) return;
  const int m = gid & 7;
  const int q = (gid >> 3) & (LQ - 1);
  const int n = gid >> 15;

  const int Tl[4] = {16, 8, 4, 2};
  const int Hl[4] = {64, 32, 16, 8};
  const int Wl[4] = {64, 32, 16, 8};
  const int st[4] = {0, 65536, 73728, 74752};

  const size_t nq = (size_t)n * LQ + q;
  const float* rp = refp + nq * LL * 3;
  const float* op = off  + (nq * MH + m) * LL * PP * 3;
  const float* ap = attn + (nq * MH + m) * LL * PP;
  const float* vbase = value + (size_t)n * LIN * CDIM + m * DH + lane;

  float acc = 0.f;
  for (int l = 0; l < LL; ++l) {
    const int Ti = Tl[l], Hi = Hl[l], Wi = Wl[l];
    const float Tf = (float)Ti, Hf = (float)Hi, Wf = (float)Wi;
    const float r0 = rp[l * 3 + 0], r1 = rp[l * 3 + 1], r2 = rp[l * 3 + 2];
    const float* vlev = vbase + (size_t)st[l] * CDIM;
    for (int p = 0; p < PP; ++p) {
      const float* o3 = op + (l * PP + p) * 3;
      const float  aw = ap[l * PP + p];
      // normalizer = (T, W, H); grid comp0->x(W), comp1->y(H), comp2->z(T)
      const float gx = 2.f * (r0 + o3[0] / Tf) - 1.f;
      const float gy = 2.f * (r1 + o3[1] / Wf) - 1.f;
      const float gz = 2.f * (r2 + o3[2] / Hf) - 1.f;
      const float x = ((gx + 1.f) * Wf - 1.f) * 0.5f;
      const float y = ((gy + 1.f) * Hf - 1.f) * 0.5f;
      const float z = ((gz + 1.f) * Tf - 1.f) * 0.5f;
      const float x0f = floorf(x), y0f = floorf(y), z0f = floorf(z);
      const float fx = x - x0f, fy = y - y0f, fz = z - z0f;
      const int x0 = (int)x0f, y0 = (int)y0f, z0 = (int)z0f;
      float samp = 0.f;
#pragma unroll
      for (int dz = 0; dz < 2; ++dz) {
        const int zi = z0 + dz; const float wz = dz ? fz : 1.f - fz;
#pragma unroll
        for (int dy = 0; dy < 2; ++dy) {
          const int yi = y0 + dy; const float wy = dy ? fy : 1.f - fy;
#pragma unroll
          for (int dx = 0; dx < 2; ++dx) {
            const int xi = x0 + dx; const float wx = dx ? fx : 1.f - fx;
            if (xi >= 0 && xi < Wi && yi >= 0 && yi < Hi && zi >= 0 && zi < Ti) {
              const int idx = (zi * Hi + yi) * Wi + xi;
              samp += wz * wy * wx * vlev[(size_t)idx * CDIM];
            }
          }
        }
      }
      acc += aw * samp;
    }
  }
  t2f16[(nq * MH + m) * DH + lane] = (_Float16)acc;
}

// ---------------------------------------------------------------------------
extern "C" void kernel_launch(void* const* d_in, const int* in_sizes, int n_in,
                              void* d_out, int out_size, void* d_ws, size_t ws_size,
                              hipStream_t stream) {
  const float* query  = (const float*)d_in[0];
  const float* refp   = (const float*)d_in[1];
  const float* inpf   = (const float*)d_in[2];
  // d_in[3]=shapes, d_in[4]=level_start (ints) — compile-time constants here
  const float* W_off  = (const float*)d_in[5];
  const float* b_off  = (const float*)d_in[6];
  const float* W_attn = (const float*)d_in[7];
  const float* b_attn = (const float*)d_in[8];
  const float* W_val  = (const float*)d_in[9];
  const float* b_val  = (const float*)d_in[10];
  const float* W_out  = (const float*)d_in[11];
  const float* b_out  = (const float*)d_in[12];
  float* out = (float*)d_out;

  // --- workspace carve (≈256 MB total) ---
  char* ws = (char*)d_ws;
  size_t cur = 0;
  auto carve = [&](size_t bytes) -> void* {
    void* p = ws + cur;
    cur = (cur + bytes + 255) & ~(size_t)255;
    return p;
  };
  const size_t nA   = (size_t)NB * LIN * CDIM;       // 38,338,560
  const size_t nQ   = (size_t)NB * LQ * CDIM;        //  2,097,152
  _Float16* Af16    = (_Float16*)carve(nA * 2);
  _Float16* Qf16    = (_Float16*)carve(nQ * 2);
  _Float16* Wt_val  = (_Float16*)carve(256 * 256 * 2);
  _Float16* Wt_off  = (_Float16*)carve(384 * 256 * 2);
  _Float16* Wt_attn = (_Float16*)carve(128 * 256 * 2);
  _Float16* Wt_out  = (_Float16*)carve(256 * 256 * 2);
  float*    valbuf  = (float*)carve(nA * 4);                       // (N,Lin,M,Dh)
  float*    offbuf  = (float*)carve((size_t)NB * LQ * 384 * 4);    // (N,Lq,M,L,P,3)
  float*    awbuf   = (float*)carve((size_t)NB * LQ * 128 * 4);    // (N,Lq,M,L*P)
  _Float16* t2f16   = (_Float16*)carve(nQ * 2);                    // (N,Lq,M,Dh)

  // 1) convert activations to f16
  f32_to_f16_kernel<<<4096, 256, 0, stream>>>(inpf, Af16, (int)nA);
  f32_to_f16_kernel<<<2048, 256, 0, stream>>>(query, Qf16, (int)nQ);

  // 2) transpose-convert weights to (Ncols, K) f16
  wt_transpose_kernel<<<(256 * 256 + 255) / 256, 256, 0, stream>>>(W_val,  Wt_val,  256, 256);
  wt_transpose_kernel<<<(256 * 384 + 255) / 256, 256, 0, stream>>>(W_off,  Wt_off,  256, 384);
  wt_transpose_kernel<<<(256 * 128 + 255) / 256, 256, 0, stream>>>(W_attn, Wt_attn, 256, 128);
  wt_transpose_kernel<<<(256 * 256 + 255) / 256, 256, 0, stream>>>(W_out,  Wt_out,  256, 256);

  // 3) value projection: (149760,256)x(256,256) — dominant GEMM
  {
    dim3 grid((NB * LIN / 32) / 4, 256 / 32);   // 1170 x 8
    gemm_f16_wmma_kernel<<<grid, 128, 0, stream>>>(Af16, Wt_val, b_val, valbuf,
                                                   NB * LIN, 256, 256);
  }
  // 4) offsets: (8192,256)x(256,384)
  {
    dim3 grid((NB * LQ / 32) / 4, 384 / 32);    // 64 x 12
    gemm_f16_wmma_kernel<<<grid, 128, 0, stream>>>(Qf16, Wt_off, b_off, offbuf,
                                                   NB * LQ, 384, 256);
  }
  // 5) attention logits: (8192,256)x(256,128)
  {
    dim3 grid((NB * LQ / 32) / 4, 128 / 32);    // 64 x 4
    gemm_f16_wmma_kernel<<<grid, 128, 0, stream>>>(Qf16, Wt_attn, b_attn, awbuf,
                                                   NB * LQ, 128, 256);
  }
  // 6) in-place softmax over NB*LQ*MH rows of 32 floats
  softmax32_kernel<<<(NB * LQ * MH) / 8, 256, 0, stream>>>(awbuf, NB * LQ * MH);

  // 7) trilinear sampling + weighted sum -> f16 heads
  msdeform_sample_kernel<<<(NB * LQ * MH) / 4, 128, 0, stream>>>(refp, offbuf, awbuf,
                                                                 valbuf, t2f16);
  // 8) output projection: (8192,256)x(256,256) -> d_out (fp32)
  {
    dim3 grid((NB * LQ / 32) / 4, 256 / 32);    // 64 x 8
    gemm_f16_wmma_kernel<<<grid, 128, 0, stream>>>(t2f16, Wt_out, b_out, out,
                                                   NB * LQ, 256, 256);
  }
}
